// BCGTransformer_52055003627697
// MI455X (gfx1250) — matrix operand order repaired and checked by reading, not compiled, not run on hardware
//
#include <hip/hip_runtime.h>
#include <hip/hip_bf16.h>

#define N_NODES 20000
#define FDIM    20
#define DDIM    256
#define HHEADS  8
#define DKDIM   32
#define LLAYERS 3
#define SSTAGES 4
#define EEDGES  320000

typedef __bf16 bf16;
typedef __attribute__((ext_vector_type(16))) __bf16 v16bf;
typedef __attribute__((ext_vector_type(8)))  __bf16 v8bf;
typedef __attribute__((ext_vector_type(8)))  float  v8f;

// ---------------------------------------------------------------- helpers
__device__ inline unsigned f2key(float f) {
    unsigned u = __float_as_uint(f);
    return (u & 0x80000000u) ? ~u : (u | 0x80000000u);
}
__device__ inline float key2f(unsigned k) {
    unsigned u = (k & 0x80000000u) ? (k & 0x7fffffffu) : ~k;
    return __uint_as_float(u);
}

// ---------------------------------------------------------------- fills / converts
__global__ void k_fill_f32(float* __restrict__ p, int n, float v) {
    int t = blockIdx.x * blockDim.x + threadIdx.x;
    if (t < n) p[t] = v;
}
__global__ void k_fill_u32(unsigned* __restrict__ p, int n, unsigned v) {
    int t = blockIdx.x * blockDim.x + threadIdx.x;
    if (t < n) p[t] = v;
}
__global__ void k_f32_to_bf16(const float* __restrict__ in, bf16* __restrict__ out, int n) {
    int t = blockIdx.x * blockDim.x + threadIdx.x;
    if (t < n) out[t] = (bf16)in[t];
}
// In f32 [K][Nc] row-major  ->  Out bf16 [Nc][K] (transposed, for WMMA B operand)
__global__ void k_transpose_bf16(const float* __restrict__ in, bf16* __restrict__ out,
                                 int K, int Nc) {
    int t = blockIdx.x * blockDim.x + threadIdx.x;
    if (t >= K * Nc) return;
    int k = t / Nc, n = t % Nc;
    out[(size_t)n * K + k] = (bf16)in[(size_t)k * Nc + n];
}

// ---------------------------------------------------------------- input projection (K=20, scalar)
__global__ __launch_bounds__(DDIM)
void k_in_proj(const float* __restrict__ x, const float* __restrict__ in_w,
               const float* __restrict__ in_b, const float* __restrict__ stage_emb,
               int s, float* __restrict__ h32, bf16* __restrict__ hbf) {
    __shared__ float xr[FDIM];
    int n = blockIdx.x, d = threadIdx.x;
    if (d < FDIM) xr[d] = x[(size_t)n * FDIM + d];
    __syncthreads();
    float acc = in_b[d] + stage_emb[(size_t)s * DDIM + d];
#pragma unroll
    for (int f = 0; f < FDIM; ++f) acc += xr[f] * in_w[(size_t)f * DDIM + d];
    h32[(size_t)n * DDIM + d] = acc;
    hbf[(size_t)n * DDIM + d] = (bf16)acc;
}

// ---------------------------------------------------------------- WMMA bf16 GEMM
// C[M][Nc] = A[M][K] @ Bt[Nc][K]^T  (+bias)  [optional exact GELU]
// wave tile: 32 rows x 64 cols (8x v_wmma_f32_16x16x32_bf16 per k-step;
// two A fragments share four B fragments -> 12 B128 loads per 8 WMMAs).
// block: 4 waves -> 32 x 256 tile.  Requires M%32==0, Nc%256==0, K%32==0.
template<bool GELU, bool F32OUT, bool BF16OUT>
__global__ __launch_bounds__(128)
void k_gemm_bf16(const bf16* __restrict__ A, const bf16* __restrict__ Bt,
                 const float* __restrict__ bias,
                 float* __restrict__ C, bf16* __restrict__ Cb,
                 int M, int Nc, int K) {
    const int lane = threadIdx.x & 31;
    const int wave = threadIdx.x >> 5;
    const int m0 = blockIdx.y * 32;
    const int n_base = blockIdx.x * 256 + wave * 64;
    const int l16 = lane & 15;
    const int kh  = lane >> 4;

    const bf16* arow0 = A  + (size_t)(m0 +  0 + l16) * K;
    const bf16* arow1 = A  + (size_t)(m0 + 16 + l16) * K;
    const bf16* b0row = Bt + (size_t)(n_base +  0 + l16) * K;
    const bf16* b1row = Bt + (size_t)(n_base + 16 + l16) * K;
    const bf16* b2row = Bt + (size_t)(n_base + 32 + l16) * K;
    const bf16* b3row = Bt + (size_t)(n_base + 48 + l16) * K;

    v8f acc00 = {}, acc01 = {}, acc02 = {}, acc03 = {};
    v8f acc10 = {}, acc11 = {}, acc12 = {}, acc13 = {};

    for (int k0 = 0; k0 < K; k0 += 32) {
        // A 16x32 bf16 fragment: lanes 0-15 hold K=[k0..k0+7]&[k0+16..k0+23],
        // lanes 16-31 hold K=[k0+8..k0+15]&[k0+24..k0+31] (ISA 7.12.2 layout).
        auto loada = [&](const bf16* ar) -> v16bf {
            v16bf a;
            v8bf lo = *(const v8bf*)(ar + k0 + 8 * kh);
            v8bf hi = *(const v8bf*)(ar + k0 + 16 + 8 * kh);
#pragma unroll
            for (int i = 0; i < 8; ++i) { a[i] = lo[i]; a[i + 8] = hi[i]; }
            return a;
        };
        // B 32x16 bf16 fragment: lane holds column n, same K split as A.
        auto loadb = [&](const bf16* br) -> v16bf {
            v16bf b;
            v8bf lo = *(const v8bf*)(br + k0 + 8 * kh);
            v8bf hi = *(const v8bf*)(br + k0 + 16 + 8 * kh);
#pragma unroll
            for (int i = 0; i < 8; ++i) { b[i] = lo[i]; b[i + 8] = hi[i]; }
            return b;
        };
        v16bf a0 = loada(arow0), a1 = loada(arow1);
        v16bf b0 = loadb(b0row), b1 = loadb(b1row), b2 = loadb(b2row), b3 = loadb(b3row);

        acc00 = __builtin_amdgcn_wmma_f32_16x16x32_bf16(false, a0, false, b0, (short)0, acc00, false, false);
        acc01 = __builtin_amdgcn_wmma_f32_16x16x32_bf16(false, a0, false, b1, (short)0, acc01, false, false);
        acc02 = __builtin_amdgcn_wmma_f32_16x16x32_bf16(false, a0, false, b2, (short)0, acc02, false, false);
        acc03 = __builtin_amdgcn_wmma_f32_16x16x32_bf16(false, a0, false, b3, (short)0, acc03, false, false);
        acc10 = __builtin_amdgcn_wmma_f32_16x16x32_bf16(false, a1, false, b0, (short)0, acc10, false, false);
        acc11 = __builtin_amdgcn_wmma_f32_16x16x32_bf16(false, a1, false, b1, (short)0, acc11, false, false);
        acc12 = __builtin_amdgcn_wmma_f32_16x16x32_bf16(false, a1, false, b2, (short)0, acc12, false, false);
        acc13 = __builtin_amdgcn_wmma_f32_16x16x32_bf16(false, a1, false, b3, (short)0, acc13, false, false);
    }

    // C/D 16x16 f32 layout: VGPR r -> (M = mbase + 8*kh + r, N = n0 + lane%16)
    auto emit = [&](v8f& acc, int mbase, int noff) {
        int n = n_base + noff + l16;
        int mb = mbase + kh * 8;
        float bv = bias ? bias[n] : 0.f;
#pragma unroll
        for (int r = 0; r < 8; ++r) {
            float v = acc[r] + bv;
            if (GELU) v = 0.5f * v * (1.f + erff(v * 0.70710678118654752f));
            size_t idx = (size_t)(mb + r) * Nc + n;
            if (F32OUT)  C[idx]  = v;
            if (BF16OUT) Cb[idx] = (bf16)v;
        }
    };
    emit(acc00, m0,      0); emit(acc01, m0,      16); emit(acc02, m0,      32); emit(acc03, m0,      48);
    emit(acc10, m0 + 16, 0); emit(acc11, m0 + 16, 16); emit(acc12, m0 + 16, 32); emit(acc13, m0 + 16, 48);
}

// ---------------------------------------------------------------- fused residual + LayerNorm
// out = LN(y + res) * g + b ; writes f32 (optional) and bf16 (optional, with ld/col for concat)
__global__ __launch_bounds__(DDIM)
void k_ln(const float* __restrict__ y, const float* __restrict__ res,
          const float* __restrict__ g, const float* __restrict__ b,
          float* __restrict__ o32, bf16* __restrict__ obf, int ldbf, int coloff) {
    int n = blockIdx.x, d = threadIdx.x;
    float v = y[(size_t)n * DDIM + d];
    if (res) v += res[(size_t)n * DDIM + d];
    __shared__ float sm[DDIM];
    sm[d] = v; __syncthreads();
    for (int off = DDIM / 2; off > 0; off >>= 1) {
        if (d < off) sm[d] += sm[d + off];
        __syncthreads();
    }
    float mu = sm[0] * (1.f / DDIM);
    __syncthreads();
    float c = v - mu;
    sm[d] = c * c; __syncthreads();
    for (int off = DDIM / 2; off > 0; off >>= 1) {
        if (d < off) sm[d] += sm[d + off];
        __syncthreads();
    }
    float var = sm[0] * (1.f / DDIM);
    float o = c * rsqrtf(var + 1e-5f) * g[d] + b[d];
    if (o32) o32[(size_t)n * DDIM + d] = o;
    if (obf) obf[(size_t)n * ldbf + coloff + d] = (bf16)o;
}

// ---------------------------------------------------------------- edge attention (3 recompute passes)
__device__ inline float edge_attn(const float* __restrict__ Q, const float* __restrict__ Kf,
                                  int si, int di, int h, float scm, float lam_fcw) {
    const float* q = Q  + (size_t)di * DDIM + h * DKDIM;
    const float* k = Kf + (size_t)si * DDIM + h * DKDIM;
    float dot = 0.f;
#pragma unroll
    for (int i = 0; i < DKDIM; i += 4) {
        float4 qv = *(const float4*)(q + i);
        float4 kv = *(const float4*)(k + i);
        dot += qv.x * kv.x + qv.y * kv.y + qv.z * kv.z + qv.w * kv.w;
    }
    return dot * 0.17677669529663687f * scm + lam_fcw;  // 1/sqrt(32)
}

__global__ void k_edge_max(const int* __restrict__ src, const int* __restrict__ dst,
                           const int* __restrict__ sc, const float* __restrict__ fcw,
                           const float* __restrict__ Q, const float* __restrict__ Kf,
                           const float* __restrict__ lam, int l, unsigned* __restrict__ mkey) {
    int t = blockIdx.x * blockDim.x + threadIdx.x;
    if (t >= EEDGES * HHEADS) return;
    int e = t >> 3, h = t & 7;
    int si = src[e], di = dst[e];
    float attn = edge_attn(Q, Kf, si, di, h, (float)sc[e], lam[l] * fcw[e]);
    atomicMax(&mkey[di * HHEADS + h], f2key(attn));
}

__global__ void k_edge_sum(const int* __restrict__ src, const int* __restrict__ dst,
                           const int* __restrict__ sc, const float* __restrict__ fcw,
                           const float* __restrict__ Q, const float* __restrict__ Kf,
                           const float* __restrict__ lam, int l,
                           const unsigned* __restrict__ mkey, float* __restrict__ ssum) {
    int t = blockIdx.x * blockDim.x + threadIdx.x;
    if (t >= EEDGES * HHEADS) return;
    int e = t >> 3, h = t & 7;
    int si = src[e], di = dst[e];
    float attn = edge_attn(Q, Kf, si, di, h, (float)sc[e], lam[l] * fcw[e]);
    float m = key2f(mkey[di * HHEADS + h]);
    atomicAdd(&ssum[di * HHEADS + h], __expf(attn - m));
}

__global__ void k_edge_agg(const int* __restrict__ src, const int* __restrict__ dst,
                           const int* __restrict__ sc, const float* __restrict__ fcw,
                           const float* __restrict__ Q, const float* __restrict__ Kf,
                           const float* __restrict__ lam, int l,
                           const unsigned* __restrict__ mkey, const float* __restrict__ ssum,
                           const float* __restrict__ V, float* __restrict__ agg) {
    int t = blockIdx.x * blockDim.x + threadIdx.x;
    if (t >= EEDGES * HHEADS) return;
    int e = t >> 3, h = t & 7;
    int si = src[e], di = dst[e];
    float attn = edge_attn(Q, Kf, si, di, h, (float)sc[e], lam[l] * fcw[e]);
    float m = key2f(mkey[di * HHEADS + h]);
    float a = __expf(attn - m) / (ssum[di * HHEADS + h] + 1e-16f);
    const float* vp = V + (size_t)si * DDIM + h * DKDIM;
    float* ap = agg + (size_t)di * DDIM + h * DKDIM;
#pragma unroll
    for (int i = 0; i < DKDIM; i += 4) {
        float4 vv = *(const float4*)(vp + i);
        atomicAdd(ap + i + 0, a * vv.x);
        atomicAdd(ap + i + 1, a * vv.y);
        atomicAdd(ap + i + 2, a * vv.z);
        atomicAdd(ap + i + 3, a * vv.w);
    }
}

// ---------------------------------------------------------------- launch
extern "C" void kernel_launch(void* const* d_in, const int* in_sizes, int n_in,
                              void* d_out, int out_size, void* d_ws, size_t ws_size,
                              hipStream_t stream) {
    (void)in_sizes; (void)n_in; (void)out_size; (void)ws_size;
    const float* x          = (const float*)d_in[0];
    const int*   edge_index = (const int*)  d_in[1];
    const int*   sc_mask    = (const int*)  d_in[2];
    const float* fc_weights = (const float*)d_in[3];
    const float* in_w       = (const float*)d_in[4];
    const float* in_b       = (const float*)d_in[5];
    const float* stage_emb  = (const float*)d_in[6];
    const float* Wq         = (const float*)d_in[7];
    const float* Wk         = (const float*)d_in[8];
    const float* Wv         = (const float*)d_in[9];
    const float* Wo         = (const float*)d_in[10];
    const float* bo         = (const float*)d_in[11];
    const float* ln1_g      = (const float*)d_in[12];
    const float* ln1_b      = (const float*)d_in[13];
    const float* ln2_g      = (const float*)d_in[14];
    const float* ln2_b      = (const float*)d_in[15];
    const float* ffn_w1     = (const float*)d_in[16];
    const float* ffn_b1     = (const float*)d_in[17];
    const float* ffn_w2     = (const float*)d_in[18];
    const float* ffn_b2     = (const float*)d_in[19];
    const float* lam        = (const float*)d_in[20];
    const float* fus_w      = (const float*)d_in[21];
    const float* fus_b      = (const float*)d_in[22];
    const float* out_g      = (const float*)d_in[23];
    const float* out_b      = (const float*)d_in[24];
    float* out = (float*)d_out;

    const size_t ND  = (size_t)N_NODES * DDIM;       // 5.12M
    const size_t N4D = ND * 4;
    const size_t DD  = (size_t)DDIM * DDIM;          // 65536
    const size_t D4D = (size_t)DDIM * 4 * DDIM;      // 262144

    char* wsp = (char*)d_ws;
    size_t off = 0;
    auto alloc = [&](size_t bytes) -> char* {
        char* p = wsp + off;
        off = (off + bytes + 255) & ~(size_t)255;
        return p;
    };
    bf16* wtq   = (bf16*)alloc(2 * LLAYERS * DD);
    bf16* wtk   = (bf16*)alloc(2 * LLAYERS * DD);
    bf16* wtv   = (bf16*)alloc(2 * LLAYERS * DD);
    bf16* wto   = (bf16*)alloc(2 * LLAYERS * DD);
    bf16* wtw1  = (bf16*)alloc(2 * LLAYERS * D4D);
    bf16* wtw2  = (bf16*)alloc(2 * LLAYERS * D4D);
    bf16* wtfus = (bf16*)alloc(2 * SSTAGES * DD);
    float*    h32    = (float*)   alloc(4 * ND);
    bf16*     hbf    = (bf16*)    alloc(2 * ND);
    float*    q32    = (float*)   alloc(4 * ND);
    float*    k32    = (float*)   alloc(4 * ND);
    float*    v32    = (float*)   alloc(4 * ND);
    unsigned* mkey   = (unsigned*)alloc(4 * (size_t)N_NODES * HHEADS);
    float*    ssum   = (float*)   alloc(4 * (size_t)N_NODES * HHEADS);
    float*    agg32  = (float*)   alloc(4 * ND);
    bf16*     aggbf  = (bf16*)    alloc(2 * ND);
    float*    y32    = (float*)   alloc(4 * ND);
    float*    ln1o32 = (float*)   alloc(4 * ND);
    bf16*     ln1obf = (bf16*)    alloc(2 * ND);
    bf16*     ffnbf  = (bf16*)    alloc(2 * N4D);
    float*    ffn232 = (float*)   alloc(4 * ND);
    bf16*     hcatbf = (bf16*)    alloc(2 * ND * SSTAGES);
    float*    fus32  = (float*)   alloc(4 * ND);

    const int TPB = 256;
    auto blk = [&](size_t n) { return (unsigned)((n + TPB - 1) / TPB); };

    // ---- one-time per launch: transpose+convert weights to bf16 [Nc][K]
    for (int l = 0; l < LLAYERS; ++l) {
        k_transpose_bf16<<<blk(DD), TPB, 0, stream>>>(Wq + l * DD, wtq + l * DD, DDIM, DDIM);
        k_transpose_bf16<<<blk(DD), TPB, 0, stream>>>(Wk + l * DD, wtk + l * DD, DDIM, DDIM);
        k_transpose_bf16<<<blk(DD), TPB, 0, stream>>>(Wv + l * DD, wtv + l * DD, DDIM, DDIM);
        k_transpose_bf16<<<blk(DD), TPB, 0, stream>>>(Wo + l * DD, wto + l * DD, DDIM, DDIM);
        k_transpose_bf16<<<blk(D4D), TPB, 0, stream>>>(ffn_w1 + l * D4D, wtw1 + l * D4D, DDIM, 4 * DDIM);
        k_transpose_bf16<<<blk(D4D), TPB, 0, stream>>>(ffn_w2 + l * D4D, wtw2 + l * D4D, 4 * DDIM, DDIM);
    }
    k_transpose_bf16<<<blk(SSTAGES * DD), TPB, 0, stream>>>(fus_w, wtfus, SSTAGES * DDIM, DDIM);

    const dim3 gblock(128);
    const dim3 gemmD(1, N_NODES / 32);          // Nc = 256
    const dim3 gemm4D(4, N_NODES / 32);         // Nc = 1024
    const size_t eth = (size_t)EEDGES * HHEADS;

    for (int s = 0; s < SSTAGES; ++s) {
        const int* srcp = edge_index + (size_t)s * 2 * EEDGES;
        const int* dstp = srcp + EEDGES;
        const int* scp  = sc_mask + (size_t)s * EEDGES;
        const float* fcwp = fc_weights + (size_t)s * EEDGES;

        k_in_proj<<<N_NODES, DDIM, 0, stream>>>(x, in_w, in_b, stage_emb, s, h32, hbf);

        for (int l = 0; l < LLAYERS; ++l) {
            const size_t wo = (size_t)l * DD;
            // QKV projections (bf16 WMMA, f32 out for edge math)
            k_gemm_bf16<false, true, false><<<gemmD, gblock, 0, stream>>>(
                hbf, wtq + wo, nullptr, q32, nullptr, N_NODES, DDIM, DDIM);
            k_gemm_bf16<false, true, false><<<gemmD, gblock, 0, stream>>>(
                hbf, wtk + wo, nullptr, k32, nullptr, N_NODES, DDIM, DDIM);
            k_gemm_bf16<false, true, false><<<gemmD, gblock, 0, stream>>>(
                hbf, wtv + wo, nullptr, v32, nullptr, N_NODES, DDIM, DDIM);

            // segment softmax state
            k_fill_u32<<<blk((size_t)N_NODES * HHEADS), TPB, 0, stream>>>(mkey, N_NODES * HHEADS, 0u);
            k_fill_f32<<<blk((size_t)N_NODES * HHEADS), TPB, 0, stream>>>(ssum, N_NODES * HHEADS, 0.f);
            k_fill_f32<<<blk(ND), TPB, 0, stream>>>(agg32, (int)ND, 0.f);

            k_edge_max<<<blk(eth), TPB, 0, stream>>>(srcp, dstp, scp, fcwp, q32, k32, lam, l, mkey);
            k_edge_sum<<<blk(eth), TPB, 0, stream>>>(srcp, dstp, scp, fcwp, q32, k32, lam, l, mkey, ssum);
            k_edge_agg<<<blk(eth), TPB, 0, stream>>>(srcp, dstp, scp, fcwp, q32, k32, lam, l, mkey, ssum,
                                                    v32, agg32);

            k_f32_to_bf16<<<blk(ND), TPB, 0, stream>>>(agg32, aggbf, (int)ND);

            // out = agg @ Wo + bo
            k_gemm_bf16<false, true, false><<<gemmD, gblock, 0, stream>>>(
                aggbf, wto + wo, bo + l * DDIM, y32, nullptr, N_NODES, DDIM, DDIM);
            // out = LN(out + h)
            k_ln<<<N_NODES, DDIM, 0, stream>>>(y32, h32, ln1_g + l * DDIM, ln1_b + l * DDIM,
                                               ln1o32, ln1obf, DDIM, 0);
            // f = gelu(out @ w1 + b1)  (bf16-only output, fused GELU)
            k_gemm_bf16<true, false, true><<<gemm4D, gblock, 0, stream>>>(
                ln1obf, wtw1 + (size_t)l * D4D, ffn_b1 + l * 4 * DDIM,
                nullptr, ffnbf, N_NODES, 4 * DDIM, DDIM);
            // f = f @ w2 + b2
            k_gemm_bf16<false, true, false><<<gemmD, gblock, 0, stream>>>(
                ffnbf, wtw2 + (size_t)l * D4D, ffn_b2 + l * DDIM,
                ffn232, nullptr, N_NODES, DDIM, 4 * DDIM);
            // h = LN(out + f) ; last layer writes bf16 straight into concat buffer
            bool last = (l == LLAYERS - 1);
            k_ln<<<N_NODES, DDIM, 0, stream>>>(ffn232, ln1o32, ln2_g + l * DDIM, ln2_b + l * DDIM,
                                               h32,
                                               last ? hcatbf : hbf,
                                               last ? SSTAGES * DDIM : DDIM,
                                               last ? s * DDIM : 0);
        }
    }

    // fusion: [N, S*D] @ fus_w + fus_b, then final LN -> d_out
    k_gemm_bf16<false, true, false><<<gemmD, gblock, 0, stream>>>(
        hcatbf, wtfus, fus_b, fus32, nullptr, N_NODES, DDIM, SSTAGES * DDIM);
    k_ln<<<N_NODES, DDIM, 0, stream>>>(fus32, nullptr, out_g, out_b, out, nullptr, 0, 0);
}
